// ReLULSTMClassifier_63230508532485
// MI455X (gfx1250) — matrix operand order repaired and checked by reading, not compiled, not run on hardware
//
#include <hip/hip_runtime.h>

typedef __attribute__((ext_vector_type(16))) __bf16 v16bf;
typedef __attribute__((ext_vector_type(8)))  float  v8f;
typedef __attribute__((ext_vector_type(4)))  __bf16 v4bf;
typedef __attribute__((ext_vector_type(4)))  int    v4i_;

#define B_   128
#define S_   1024
#define I_   256
#define H_   512
#define O_   10

#define THREADS 128        // 4 waves
#define WG_M   32          // batch rows per workgroup
#define WG_NH  16          // hidden columns per workgroup
#define WG_NG  64          // gate columns per workgroup (4 gates x WG_NH)
#define NWG_M  4
#define NWG_N  32
#define NUM_WG (NWG_M * NWG_N)

#define PAD    8
#define LDH    (H_ + PAD)  // 520 elems; 1040B row stride = 65*16 (16B aligned, bank shift 4 dw)
#define LDI    (I_ + PAD)  // 264 elems; 528B row stride  = 33*16

#if __has_builtin(__builtin_amdgcn_global_load_async_to_lds_b128)
#define ASYNC_BUILTIN 1
#endif

__device__ __forceinline__ void async_copy16(const __bf16* g, __bf16* l) {
#ifdef ASYNC_BUILTIN
  __attribute__((address_space(1))) v4i_* gp =
      (__attribute__((address_space(1))) v4i_*)((v4i_*)const_cast<__bf16*>(g));
  __attribute__((address_space(3))) v4i_* lp =
      (__attribute__((address_space(3))) v4i_*)((v4i_*)l);
  __builtin_amdgcn_global_load_async_to_lds_b128(gp, lp, 0, 0);
#else
  unsigned lofs = (unsigned)(uintptr_t)l;
  unsigned long long ga = (unsigned long long)(uintptr_t)g;
  asm volatile("global_load_async_to_lds_b128 %0, %1, off"
               :: "v"(lofs), "v"(ga) : "memory");
#endif
}

__device__ __forceinline__ void wait_async0() {
#if __has_builtin(__builtin_amdgcn_s_wait_asynccnt)
  __builtin_amdgcn_s_wait_asynccnt(0);
#else
  asm volatile("s_wait_asynccnt 0" ::: "memory");
#endif
}

__device__ __forceinline__ float sigmoidf_(float x) {
  return 1.0f / (1.0f + __expf(-x));
}

__device__ __forceinline__ void gridBarrier(unsigned* cnt, unsigned* gen) {
  __syncthreads();
  if (threadIdx.x == 0) {
    __threadfence();
    unsigned my = __hip_atomic_load(gen, __ATOMIC_RELAXED, __HIP_MEMORY_SCOPE_AGENT);
    unsigned prev = __hip_atomic_fetch_add(cnt, 1u, __ATOMIC_ACQ_REL, __HIP_MEMORY_SCOPE_AGENT);
    if (prev == NUM_WG - 1) {
      __hip_atomic_store(cnt, 0u, __ATOMIC_RELAXED, __HIP_MEMORY_SCOPE_AGENT);
      __hip_atomic_fetch_add(gen, 1u, __ATOMIC_RELEASE, __HIP_MEMORY_SCOPE_AGENT);
    } else {
      while (__hip_atomic_load(gen, __ATOMIC_ACQUIRE, __HIP_MEMORY_SCOPE_AGENT) == my) {
        __builtin_amdgcn_s_sleep(1);
      }
    }
    __threadfence();
  }
  __syncthreads();
}

// ---------------- x pre-conversion: fp32 -> bf16 (fully parallel, one pass) ---
extern "C" __global__ __launch_bounds__(256)
void x_to_bf16(const float* __restrict__ x, unsigned short* __restrict__ xb) {
  size_t i = ((size_t)blockIdx.x * 256 + threadIdx.x) * 4;
  float4 v = *(const float4*)(x + i);
  v4bf p; p[0] = (__bf16)v.x; p[1] = (__bf16)v.y; p[2] = (__bf16)v.z; p[3] = (__bf16)v.w;
  *(v4bf*)((__bf16*)xb + i) = p;
}

// ---------------- persistent recurrent kernel -------------------------------
extern "C" __global__ __launch_bounds__(THREADS)
void relu_lstm_persistent(const float* __restrict__ W_ih,
                          const float* __restrict__ W_hh,
                          const float* __restrict__ b_ih,
                          const float* __restrict__ b_hh,
                          const float* __restrict__ fc_w,
                          const float* __restrict__ fc_b,
                          const unsigned short* __restrict__ xb_us,
                          float* __restrict__ out,
                          unsigned short* __restrict__ h0_us,
                          unsigned short* __restrict__ h1_us,
                          unsigned* __restrict__ barCnt,
                          unsigned* __restrict__ barGen)
{
  __shared__ __align__(16) __bf16 sWhh[WG_NG * LDH];  // 66560 B
  __shared__ __align__(16) __bf16 sWih[WG_NG * LDI];  // 33792 B
  __shared__ __align__(16) __bf16 sHA [WG_M * LDH];   // 33280 B
  __shared__ __align__(16) __bf16 sXA [WG_M * LDI];   // 16896 B
  __shared__ __align__(16) float  sG  [WG_M * WG_NG]; //  8192 B
  __shared__             float  sBias[WG_NG];         //   256 B

  const __bf16* xbf = reinterpret_cast<const __bf16*>(xb_us);
  __bf16* h0buf = reinterpret_cast<__bf16*>(h0_us);
  __bf16* h1buf = reinterpret_cast<__bf16*>(h1_us);

  const int tid  = threadIdx.x;
  const int nb   = blockIdx.x;     // 0..31 : hidden-column chunk
  const int mb   = blockIdx.y;     // 0..3  : batch-row block
  const int row0 = mb * WG_M;

  // ---- prefetch x(t=0) tile into LDS via async copies (overlaps weight preload)
  for (int c = tid; c < (WG_M * I_) / 8; c += THREADS) {
    int e = c * 8; int r = e >> 8; int k = e & (I_ - 1);
    async_copy16(xbf + ((size_t)(row0 + r) * S_ + 0) * I_ + k, sXA + r * LDI + k);
  }

  // ---- one-time weight preload into LDS (fp32 -> bf16) ----
  // local gate row r: gate g = r/16, j = r%16 -> global gate row g*H + nb*16 + j
  for (int i = tid; i < WG_NG * (H_ / 4); i += THREADS) {
    int r  = i / (H_ / 4);
    int kq = (i % (H_ / 4)) * 4;
    int g = r >> 4, j = r & 15;
    int grow = g * H_ + nb * WG_NH + j;
    float4 w = *(const float4*)(W_hh + (size_t)grow * H_ + kq);
    v4bf p; p[0] = (__bf16)w.x; p[1] = (__bf16)w.y; p[2] = (__bf16)w.z; p[3] = (__bf16)w.w;
    *(v4bf*)(sWhh + r * LDH + kq) = p;
  }
  for (int i = tid; i < WG_NG * (I_ / 4); i += THREADS) {
    int r  = i / (I_ / 4);
    int kq = (i % (I_ / 4)) * 4;
    int g = r >> 4, j = r & 15;
    int grow = g * H_ + nb * WG_NH + j;
    float4 w = *(const float4*)(W_ih + (size_t)grow * I_ + kq);
    v4bf p; p[0] = (__bf16)w.x; p[1] = (__bf16)w.y; p[2] = (__bf16)w.z; p[3] = (__bf16)w.w;
    *(v4bf*)(sWih + r * LDI + kq) = p;
  }
  for (int i = tid; i < WG_NG; i += THREADS) {
    int g = i >> 4, j = i & 15;
    int col = g * H_ + nb * WG_NH + j;
    sBias[i] = b_ih[col] + b_hh[col];
  }

  // cell state in registers: elements (tid*4 + q) of [32 rows][16 cols]
  float cc[4] = {0.f, 0.f, 0.f, 0.f};

  const int lane  = tid & 31;
  const int wave  = tid >> 5;    // 0..3
  const int nt    = wave;        // N tile (16 gate cols)
  const int lhalf = lane >> 4;   // K-half selector
  const int lmod  = lane & 15;

  for (int t = 0; t < S_; ++t) {
    const __bf16* hrd = (t & 1) ? h1buf : h0buf;
    __bf16*       hwr = (t & 1) ? h0buf : h1buf;

    // ---- stage h rows of this M-block: 32x512 bf16, async global->LDS ----
    for (int c = tid; c < (WG_M * H_) / 8; c += THREADS) {
      int e = c * 8; int r = e >> 9; int k = e & (H_ - 1);
      async_copy16(hrd + (size_t)(row0 + r) * H_ + k, sHA + r * LDH + k);
    }
    wait_async0();     // h(t) and x(t) tiles resident
    __syncthreads();

    // ---- x-part GEMM first (so we can prefetch x(t+1) during h-part) ----
    v8f acc0 = {0.f,0.f,0.f,0.f,0.f,0.f,0.f,0.f};
    v8f acc1 = {0.f,0.f,0.f,0.f,0.f,0.f,0.f,0.f};
    {
      const int arow0 = lmod * LDI;
      const int arow1 = (16 + lmod) * LDI;
      const int brow  = (nt * 16 + lmod) * LDI;
      #pragma unroll 4
      for (int k0 = 0; k0 < I_; k0 += 32) {
        union { uint4 u[2]; v16bf v; } ta0, ta1, tb;
        tb.u[0]  = *(const uint4*)(sWih + brow  + k0 + lhalf * 8);
        tb.u[1]  = *(const uint4*)(sWih + brow  + k0 + 16 + lhalf * 8);
        ta0.u[0] = *(const uint4*)(sXA  + arow0 + k0 + lhalf * 8);
        ta0.u[1] = *(const uint4*)(sXA  + arow0 + k0 + 16 + lhalf * 8);
        ta1.u[0] = *(const uint4*)(sXA  + arow1 + k0 + lhalf * 8);
        ta1.u[1] = *(const uint4*)(sXA  + arow1 + k0 + 16 + lhalf * 8);
        acc0 = __builtin_amdgcn_wmma_f32_16x16x32_bf16(false, ta0.v, false, tb.v,
                                                       (short)0, acc0, false, false);
        acc1 = __builtin_amdgcn_wmma_f32_16x16x32_bf16(false, ta1.v, false, tb.v,
                                                       (short)0, acc1, false, false);
      }
    }
    __syncthreads();   // everyone done reading sXA -> safe to overwrite

    // ---- issue async prefetch of x(t+1) into sXA; overlaps h-part GEMM ----
    if (t + 1 < S_) {
      for (int c = tid; c < (WG_M * I_) / 8; c += THREADS) {
        int e = c * 8; int r = e >> 8; int k = e & (I_ - 1);
        async_copy16(xbf + ((size_t)(row0 + r) * S_ + (t + 1)) * I_ + k,
                     sXA + r * LDI + k);
      }
    }

    // ---- h-part GEMM: K = 512, B fragment shared across both M tiles ----
    {
      const int arow0 = lmod * LDH;
      const int arow1 = (16 + lmod) * LDH;
      const int brow  = (nt * 16 + lmod) * LDH;
      #pragma unroll 4
      for (int k0 = 0; k0 < H_; k0 += 32) {
        union { uint4 u[2]; v16bf v; } ta0, ta1, tb;
        tb.u[0]  = *(const uint4*)(sWhh + brow  + k0 + lhalf * 8);
        tb.u[1]  = *(const uint4*)(sWhh + brow  + k0 + 16 + lhalf * 8);
        ta0.u[0] = *(const uint4*)(sHA  + arow0 + k0 + lhalf * 8);
        ta0.u[1] = *(const uint4*)(sHA  + arow0 + k0 + 16 + lhalf * 8);
        ta1.u[0] = *(const uint4*)(sHA  + arow1 + k0 + lhalf * 8);
        ta1.u[1] = *(const uint4*)(sHA  + arow1 + k0 + 16 + lhalf * 8);
        acc0 = __builtin_amdgcn_wmma_f32_16x16x32_bf16(false, ta0.v, false, tb.v,
                                                       (short)0, acc0, false, false);
        acc1 = __builtin_amdgcn_wmma_f32_16x16x32_bf16(false, ta1.v, false, tb.v,
                                                       (short)0, acc1, false, false);
      }
    }

    // ---- spill accumulators (C layout: VGPR i -> M = i + 8*lhalf) ----
    #pragma unroll
    for (int i = 0; i < 8; ++i) {
      int r0 = lhalf * 8 + i;
      sG[r0 * WG_NG + nt * 16 + lmod]        = acc0[i];
      sG[(16 + r0) * WG_NG + nt * 16 + lmod] = acc1[i];
    }
    __syncthreads();

    // ---- gate combine + cell update; packed bf16x4 h store ----
    {
      int e0 = tid * 4;
      int r  = e0 >> 4;          // 0..31
      int j0 = e0 & 15;          // multiple of 4
      const float* grow = sG + r * WG_NG;
      float hv[4];
      #pragma unroll
      for (int q = 0; q < 4; ++q) {
        int j = j0 + q;
        float yi = grow[j]      + sBias[j];
        float yf = grow[16 + j] + sBias[16 + j];
        float yg = grow[32 + j] + sBias[32 + j];
        float yo = grow[48 + j] + sBias[48 + j];
        float ig = sigmoidf_(yi);
        float fg = sigmoidf_(yf);
        float gg = fmaxf(yg, 0.0f);
        float og = sigmoidf_(yo);
        cc[q] = fg * cc[q] + ig * gg;
        hv[q] = og * cc[q];
      }
      v4bf hp; hp[0] = (__bf16)hv[0]; hp[1] = (__bf16)hv[1];
               hp[2] = (__bf16)hv[2]; hp[3] = (__bf16)hv[3];
      *(v4bf*)(hwr + (size_t)(row0 + r) * H_ + nb * WG_NH + j0) = hp;
    }

    gridBarrier(barCnt, barGen);   // h(t) visible device-wide before t+1 reads
  }

  // ---- final FC head: out = h @ fc_w.T + fc_b ; final h is in h0buf ----
  if (nb == 0) {
    const __bf16* hfin = h0buf;
    for (int e = tid; e < WG_M * O_; e += THREADS) {
      int r = e / O_, o = e % O_;
      const __bf16* hp = hfin + (size_t)(row0 + r) * H_;
      const float*  wp = fc_w + o * H_;
      float s = fc_b[o];
      for (int k = 0; k < H_; ++k) s += (float)hp[k] * wp[k];
      out[(row0 + r) * O_ + o] = s;
    }
  }
}

extern "C" void kernel_launch(void* const* d_in, const int* in_sizes, int n_in,
                              void* d_out, int out_size, void* d_ws, size_t ws_size,
                              hipStream_t stream) {
  (void)in_sizes; (void)n_in; (void)out_size; (void)ws_size;
  const float* x    = (const float*)d_in[0];
  const float* W_ih = (const float*)d_in[1];
  const float* W_hh = (const float*)d_in[2];
  const float* b_ih = (const float*)d_in[3];
  const float* b_hh = (const float*)d_in[4];
  const float* fc_w = (const float*)d_in[5];
  const float* fc_b = (const float*)d_in[6];

  unsigned char* ws = (unsigned char*)d_ws;
  unsigned* barCnt = (unsigned*)(ws + 0);
  unsigned* barGen = (unsigned*)(ws + 64);
  unsigned short* h0 = (unsigned short*)(ws + 256);
  unsigned short* h1 = (unsigned short*)(ws + 256 + (size_t)B_ * H_ * 2);
  unsigned short* xb = (unsigned short*)(ws + 256 + 2 * (size_t)B_ * H_ * 2);
  size_t initBytes = 256 + 2 * (size_t)B_ * H_ * 2;   // barrier state + both h buffers

  (void)hipMemsetAsync(d_ws, 0, initBytes, stream);

  // x: fp32 -> bf16 (one pass, fully parallel)
  int nconv = (B_ * S_ * I_) / 4;                      // float4 per thread
  x_to_bf16<<<nconv / 256, 256, 0, stream>>>(x, xb);

  dim3 grid(NWG_N, NWG_M, 1);
  relu_lstm_persistent<<<grid, THREADS, 0, stream>>>(
      W_ih, W_hh, b_ih, b_hh, fc_w, fc_b, xb,
      (float*)d_out, h0, h1, barCnt, barGen);
}
